// LSTM_73804718014748
// MI455X (gfx1250) — matrix operand order repaired and checked by reading, not compiled, not run on hardware
//
#include <hip/hip_runtime.h>
#include <stddef.h>

// ---------------------------------------------------------------------------
// LSTM on MI455X (gfx1250), bf16 WMMA, persistent recurrent kernel.
//   B=64, T=512, I=256, H=1024, O=256
// Recurrent step t:  pre[g] = [x_t ; h_{t-1}] @ [Wx_g ; Wh_g]^T + b_g
//   => GEMM M=64, N=4096 (4 gates x 1024), K=1280 (8 x-Ktiles + 32 h-Ktiles)
// ---------------------------------------------------------------------------

typedef __bf16 v16bf __attribute__((ext_vector_type(16)));
typedef float  v8f   __attribute__((ext_vector_type(8)));
typedef unsigned int v4u __attribute__((ext_vector_type(4)));
typedef int v4i __attribute__((ext_vector_type(4)));
typedef int v8i __attribute__((ext_vector_type(8)));

#define NWG       32
#define T_STEPS   512
#define KT_TOTAL  40   // 8 (x, I=256) + 32 (h, H=1024), K-tile = 32
#define KT_X      8
#define TILE_USH  512  // one 32x16 (or 16x32) bf16 fragment tile = 512 bf16 = 1KB

// workspace layout (bytes, all tile regions 1024-aligned)
#define WS_WPACK     ((size_t)0)                       // 4*64*40*512 ush = 10,485,760 B
#define WS_WOUTPACK  ((size_t)10485760)                // 16*32*512 ush  =    524,288 B
#define WS_XPACK     ((size_t)11010048)                // 512*4*8*512 ush= 16,777,216 B
#define WS_HPACK     ((size_t)27787264)                // 2*4*32*512 ush =    262,144 B
#define WS_CNT       ((size_t)28049408)                // grid-barrier counter

__device__ __forceinline__ unsigned short f32_to_bf16(float f) {
  unsigned u = __float_as_uint(f);
  unsigned r = u + 0x7FFFu + ((u >> 16) & 1u);   // round-to-nearest-even
  return (unsigned short)(r >> 16);
}

// hardware V_TANH_F32 when the builtin exists (gfx1250 TRANS op), OCML otherwise
__device__ __forceinline__ float fast_tanh(float x) {
#if __has_builtin(__builtin_amdgcn_tanhf)
  return __builtin_amdgcn_tanhf(x);
#elif __has_builtin(__builtin_amdgcn_tanh_f32)
  return __builtin_amdgcn_tanh_f32(x);
#else
  return tanhf(x);
#endif
}

__device__ __forceinline__ float sigmoidf_fast(float x) {
#if __has_builtin(__builtin_amdgcn_rcpf)
  return __builtin_amdgcn_rcpf(1.0f + __expf(-x));   // v_exp_f32 + v_rcp_f32
#else
  return 1.0f / (1.0f + __expf(-x));
#endif
}

// ---------------------------------------------------------------------------
// init: zero h ping-pong buffers (h0 = 0) and the grid-barrier counter.
// ---------------------------------------------------------------------------
__global__ void init_ws(unsigned* __restrict__ hpack_u32, unsigned* __restrict__ cnt) {
  int idx = blockIdx.x * 256 + threadIdx.x;
  if (idx < 65536) hpack_u32[idx] = 0u;      // 262,144 B of hpack
  if (idx == 65536) *cnt = 0u;
}

// ---------------------------------------------------------------------------
// Pack gate weights into bf16 B-fragment tiles.
// B tile (32x16 KxN): lane l holds N = l%16, K = (e/8)*16 + (l/16)*8 + e%8 (e=0..15),
// stored as tile*512 + l*16 + e.  Tile index = ((g*64+nt)*40+kt).
// kt<8: from Wx (K over I), else from Wh (K over H).
// ---------------------------------------------------------------------------
__global__ void pack_gate_w(const float* __restrict__ Wgx, const float* __restrict__ Wgh,
                            const float* __restrict__ Wix, const float* __restrict__ Wih,
                            const float* __restrict__ Wfx, const float* __restrict__ Wfh,
                            const float* __restrict__ Wox, const float* __restrict__ Woh,
                            unsigned short* __restrict__ wpack) {
  int idx = blockIdx.x * 256 + threadIdx.x;
  if (idx >= 4 * 64 * 40 * 512) return;
  int e    = idx & 15;
  int lane = (idx >> 4) & 31;
  int tile = idx >> 9;
  int kt = tile % 40;
  int nt = (tile / 40) & 63;
  int g  = tile / (40 * 64);
  int n  = lane & 15;
  int k  = ((e >> 3) << 4) + ((lane >> 4) << 3) + (e & 7);
  int ng = nt * 16 + n;
  const float* Wx[4] = {Wgx, Wix, Wfx, Wox};
  const float* Wh[4] = {Wgh, Wih, Wfh, Woh};
  float w = (kt < KT_X) ? Wx[g][(size_t)ng * 256 + kt * 32 + k]
                        : Wh[g][(size_t)ng * 1024 + (kt - KT_X) * 32 + k];
  wpack[idx] = f32_to_bf16(w);
}

// Wout [256 x 1024] -> 16 Ntiles x 32 Ktiles of B-fragment tiles.
__global__ void pack_wout(const float* __restrict__ Wout, unsigned short* __restrict__ wop) {
  int idx = blockIdx.x * 256 + threadIdx.x;
  if (idx >= 16 * 32 * 512) return;
  int e    = idx & 15;
  int lane = (idx >> 4) & 31;
  int tile = idx >> 9;
  int kt = tile & 31;
  int nt = tile >> 5;
  int ng = nt * 16 + (lane & 15);
  int kg = kt * 32 + ((e >> 3) << 4) + ((lane >> 4) << 3) + (e & 7);
  wop[idx] = f32_to_bf16(Wout[(size_t)ng * 1024 + kg]);
}

// ---------------------------------------------------------------------------
// Pack x into bf16 A-fragment tiles, per (t, mtile, ktile).
// A 16x32 bf16 layout: lanes 0-15 -> M=lane, K in {0..7}u{16..23};
//                      lanes 16-31 -> M=lane-16, K in {8..15}u{24..31}.
// element e = vgpr*2 + c ; vgpr = group*4 + pos/2 ; k = group*16 + half*8 + pos.
// ---------------------------------------------------------------------------
__global__ void pack_x(const float* __restrict__ x, unsigned short* __restrict__ xpack) {
  int idx = blockIdx.x * 256 + threadIdx.x;
  if (idx >= 512 * 4 * 8 * 512) return;
  int e    = idx & 15;
  int lane = (idx >> 4) & 31;
  int tile = idx >> 9;
  int kt = tile & 7;
  int mt = (tile >> 3) & 3;
  int t  = tile >> 5;
  int m    = lane & 15;
  int half = lane >> 4;
  int vg = e >> 1, cc = e & 1;
  int group = vg >> 2;
  int pos   = ((vg & 3) << 1) + cc;
  int k = (group << 4) + (half << 3) + pos;
  int b = mt * 16 + m;
  int i = kt * 32 + k;
  xpack[idx] = f32_to_bf16(x[((size_t)b * 512 + t) * 256 + i]);
}

// ---------------------------------------------------------------------------
// Persistent recurrent kernel: 32 WGs x 8 waves = 256 waves.
// Wave p = wg*8 + w  ->  mt = p & 3 (batch tile), nt = p >> 2 (H tile 0..63).
// Each wave: 4 gate accumulators (v8f) + resident cell state c (v8f).
// LDS: step-invariant recurrent B tiles for h-Ktiles 0..7 staged once by the
// Tensor Data Mover (3-D strided D#: 8KB x 2 nt x 4 gates -> contiguous 64KB),
// LDS tile order gate-major: ((g*2+ntl)*8+kth).
// ---------------------------------------------------------------------------
__global__ __launch_bounds__(256) void lstm_persistent(
    const unsigned short* __restrict__ wpack,
    const unsigned short* __restrict__ xpack,
    unsigned short* __restrict__ hpack,
    const float* __restrict__ bg, const float* __restrict__ bi,
    const float* __restrict__ bf, const float* __restrict__ bo,
    unsigned* __restrict__ barrier_cnt) {
  __shared__ __attribute__((aligned(32))) unsigned short smemB[64 * TILE_USH];  // 64 KB

  const int tid  = threadIdx.x;
  const int wave = tid >> 5;
  const int lane = tid & 31;
  const int p   = blockIdx.x * 8 + wave;
  const int mt  = p & 3;
  const int nt  = p >> 2;
  const int nt0 = blockIdx.x * 2;
  const int ntl = nt - nt0;  // 0 or 1

  // ---- stage LDS via TDM: tiles for (g=0..3, ntl=0..1, kth=0..7)
  //      src: wpack byte addr of tile ((g*64 + nt0+ntl)*40 + 8)  (8KB contiguous per (g,ntl))
  //      3-D: dim0 = 8KB (1024 x 8B), dim1 = ntl (x2, stride 40KB), dim2 = g (x4, stride 2560KB)
#if __has_builtin(__builtin_amdgcn_tensor_load_to_lds)
  if (wave == 0) {
    unsigned lds_base = (unsigned)(size_t)(&smemB[0]);
    unsigned long long ga =
        (unsigned long long)(size_t)wpack + (((size_t)nt0 * 40 + KT_X) * TILE_USH) * 2u;
    v4u g0 = { 1u,                                   // count = 1 (valid descriptor)
               lds_base,                             // lds_addr [63:32]
               (unsigned)(ga & 0xFFFFFFFFu),         // global_addr lo
               (unsigned)(ga >> 32) | (2u << 30) };  // global_addr hi | type=2 (image)
    v8i g1 = { (int)(3u << 16),        // data_size = 3 (8 bytes)
               (int)(1024u << 16),     // tensor_dim0 = 1024 (bits [79:48], lo16 here)
               (int)(2u << 16),        // tensor_dim0 hi = 0 | tensor_dim1 = 2 (lo16)
               (int)(1024u << 16),     // tensor_dim1 hi = 0 | tile_dim0 = 1024
               (int)(2u | (4u << 16)), // tile_dim1 = 2 | tile_dim2 = 4
               5120,                   // tensor_dim0_stride = 40KB/8 (lo32)
               0,                      // stride0 hi16 | tensor_dim1_stride lo16 (0x50000 & 0xFFFF = 0)
               5 };                    // tensor_dim1_stride [47:16] = 0x50000 >> 16
    v4i g2 = { 4, 0, 0, 0 };           // tensor_dim2 = 4 ; tile_dim3 = 0
    v4i g3 = { 0, 0, 0, 0 };
#if defined(__clang_major__) && (__clang_major__ >= 23)
    v8i g4 = { 0, 0, 0, 0, 0, 0, 0, 0 };
    __builtin_amdgcn_tensor_load_to_lds(g0, g1, g2, g3, g4, 0);
#else
    __builtin_amdgcn_tensor_load_to_lds(g0, g1, g2, g3, 0);
#endif
    __builtin_amdgcn_s_wait_tensorcnt(0);
  }
#else
  {
    const uint4* wsrc = (const uint4*)wpack;
    uint4* sdst = (uint4*)smemB;
    for (int i = tid; i < 64 * 64; i += 256) {   // 64 tiles x 64 uint4
      int tileId = i >> 6, chunk = i & 63;
      int g_s   = tileId >> 4;
      int ntl_s = (tileId >> 3) & 1;
      int kth_s = tileId & 7;
      int srcTile = (g_s * 64 + nt0 + ntl_s) * 40 + KT_X + kth_s;
      sdst[i] = wsrc[(size_t)srcTile * 64 + chunk];
    }
  }
#endif
  __syncthreads();

  const int ncol = nt * 16 + (lane & 15);
  const float bias0 = bg[ncol], bias1 = bi[ncol], bias2 = bf[ncol], bias3 = bo[ncol];
  const size_t GSTRIDE = (size_t)64 * 40 * TILE_USH;  // gate stride in wpack (ush)

  v8f c = {};
  unsigned phase = 1;

#pragma unroll 1
  for (int t = 0; t < T_STEPS; ++t) {
    const int rp = t & 1, wp = (t + 1) & 1;
    v8f a0, a1, a2, a3;
#pragma unroll
    for (int r = 0; r < 8; ++r) { a0[r] = bias0; a1[r] = bias1; a2[r] = bias2; a3[r] = bias3; }

    // base pointers for this step (all straight-line, no generic pointers)
    const unsigned short* xA = xpack + (((size_t)t * 4 + mt) * 8) * TILE_USH + lane * 16;
    const unsigned short* hA = hpack + (((size_t)rp * 4 + mt) * 32) * TILE_USH + lane * 16;
    const unsigned short* gB = wpack + ((size_t)nt * 40) * TILE_USH + lane * 16;

    // ---- phase 1: x contribution, B from global (kt = 0..7)
#pragma unroll 1
    for (int kt = 0; kt < KT_X; ++kt) {
      v16bf afrag = *(const v16bf*)(xA + (size_t)kt * TILE_USH);
      const unsigned short* gb = gB + (size_t)kt * TILE_USH;
      v16bf b0 = *(const v16bf*)(gb + 0 * GSTRIDE);
      v16bf b1 = *(const v16bf*)(gb + 1 * GSTRIDE);
      v16bf b2 = *(const v16bf*)(gb + 2 * GSTRIDE);
      v16bf b3 = *(const v16bf*)(gb + 3 * GSTRIDE);
      a0 = __builtin_amdgcn_wmma_f32_16x16x32_bf16(false, afrag, false, b0, (short)0, a0, false, false);
      a1 = __builtin_amdgcn_wmma_f32_16x16x32_bf16(false, afrag, false, b1, (short)0, a1, false, false);
      a2 = __builtin_amdgcn_wmma_f32_16x16x32_bf16(false, afrag, false, b2, (short)0, a2, false, false);
      a3 = __builtin_amdgcn_wmma_f32_16x16x32_bf16(false, afrag, false, b3, (short)0, a3, false, false);
    }

    // ---- phase 2: h contribution, B from LDS (h-Ktiles 0..7)
#pragma unroll 1
    for (int kth = 0; kth < 8; ++kth) {
      v16bf afrag = *(const v16bf*)(hA + (size_t)kth * TILE_USH);
      const unsigned short* sb = &smemB[((size_t)ntl * 8 + kth) * TILE_USH + lane * 16];
      v16bf b0 = *(const v16bf*)(sb + 0 * 8192);   // gate stride in LDS = 2*8*512 ush
      v16bf b1 = *(const v16bf*)(sb + 1 * 8192);
      v16bf b2 = *(const v16bf*)(sb + 2 * 8192);
      v16bf b3 = *(const v16bf*)(sb + 3 * 8192);
      a0 = __builtin_amdgcn_wmma_f32_16x16x32_bf16(false, afrag, false, b0, (short)0, a0, false, false);
      a1 = __builtin_amdgcn_wmma_f32_16x16x32_bf16(false, afrag, false, b1, (short)0, a1, false, false);
      a2 = __builtin_amdgcn_wmma_f32_16x16x32_bf16(false, afrag, false, b2, (short)0, a2, false, false);
      a3 = __builtin_amdgcn_wmma_f32_16x16x32_bf16(false, afrag, false, b3, (short)0, a3, false, false);
    }

    // ---- phase 3: h contribution, B from global (h-Ktiles 8..31, kt = 16..39)
#pragma unroll 1
    for (int kt = 16; kt < KT_TOTAL; ++kt) {
      v16bf afrag = *(const v16bf*)(hA + (size_t)(kt - KT_X) * TILE_USH);
      const unsigned short* gb = gB + (size_t)kt * TILE_USH;
      v16bf b0 = *(const v16bf*)(gb + 0 * GSTRIDE);
      v16bf b1 = *(const v16bf*)(gb + 1 * GSTRIDE);
      v16bf b2 = *(const v16bf*)(gb + 2 * GSTRIDE);
      v16bf b3 = *(const v16bf*)(gb + 3 * GSTRIDE);
      a0 = __builtin_amdgcn_wmma_f32_16x16x32_bf16(false, afrag, false, b0, (short)0, a0, false, false);
      a1 = __builtin_amdgcn_wmma_f32_16x16x32_bf16(false, afrag, false, b1, (short)0, a1, false, false);
      a2 = __builtin_amdgcn_wmma_f32_16x16x32_bf16(false, afrag, false, b2, (short)0, a2, false, false);
      a3 = __builtin_amdgcn_wmma_f32_16x16x32_bf16(false, afrag, false, b3, (short)0, a3, false, false);
    }

    // elementwise LSTM cell update; c stays in registers, h scattered out
    v8f hv;
#pragma unroll
    for (int r = 0; r < 8; ++r) {
      float gg = fast_tanh(a0[r]);
      float ii = sigmoidf_fast(a1[r]);
      float ff = sigmoidf_fast(a2[r]);
      float oo = sigmoidf_fast(a3[r]);
      float cv = gg * ii + c[r] * ff;
      c[r] = cv;
      hv[r] = fast_tanh(cv) * oo;
    }

    // write h_t in bf16 A-fragment layout into ping-pong buffer (parity wp)
#pragma unroll
    for (int r = 0; r < 8; ++r) {
      int m  = r + ((lane >> 4) << 3);        // local batch row 0..15
      int ng = nt * 16 + (lane & 15);         // global H index
      int kth = ng >> 5;                      // h K-tile
      int k   = ng & 31;
      int half  = (k >> 3) & 1;
      int laneA = m + (half << 4);
      int group = k >> 4;
      int pos   = k & 7;
      int e = (group * 4 + (pos >> 1)) * 2 + (pos & 1);
      hpack[((((size_t)wp * 4 + mt) * 32) + kth) * TILE_USH + laneA * 16 + e] = f32_to_bf16(hv[r]);
    }

    // ---- grid barrier (monotone phases; counter reset by init_ws each launch)
    __syncthreads();
    if (tid == 0) {
      __threadfence();
      atomicAdd(barrier_cnt, 1u);
      const unsigned target = (unsigned)NWG * phase;
      while (__hip_atomic_load(barrier_cnt, __ATOMIC_RELAXED, __HIP_MEMORY_SCOPE_AGENT) < target)
        __builtin_amdgcn_s_sleep(1);
    }
    __syncthreads();
    __threadfence();
    ++phase;
  }
}

// ---------------------------------------------------------------------------
// out = h_T @ Wout^T + bout   (M=64, N=256, K=1024) ; h_T is hpack parity 0.
// One wave per 16x16 output tile: 64 tiles.
// ---------------------------------------------------------------------------
__global__ __launch_bounds__(32) void lstm_out(
    const unsigned short* __restrict__ hpack,
    const unsigned short* __restrict__ woutpack,
    const float* __restrict__ bout,
    float* __restrict__ out) {
  const int lane = threadIdx.x;
  const int tile = blockIdx.x;   // 64 = 4 mt x 16 nt
  const int mt = tile & 3;
  const int nt = tile >> 2;
  const float bv = bout[nt * 16 + (lane & 15)];
  v8f acc;
#pragma unroll
  for (int r = 0; r < 8; ++r) acc[r] = bv;
#pragma unroll 1
  for (int kt = 0; kt < 32; ++kt) {
    v16bf a = *(const v16bf*)(hpack + (((size_t)mt * 32 + kt) * TILE_USH) + lane * 16);
    v16bf b = *(const v16bf*)(woutpack + (((size_t)nt * 32 + kt) * TILE_USH) + lane * 16);
    acc = __builtin_amdgcn_wmma_f32_16x16x32_bf16(false, a, false, b, (short)0, acc, false, false);
  }
#pragma unroll
  for (int r = 0; r < 8; ++r) {
    int m = mt * 16 + r + ((lane >> 4) << 3);
    int n = nt * 16 + (lane & 15);
    out[(size_t)m * 256 + n] = acc[r];
  }
}

// ---------------------------------------------------------------------------
extern "C" void kernel_launch(void* const* d_in, const int* in_sizes, int n_in,
                              void* d_out, int out_size, void* d_ws, size_t ws_size,
                              hipStream_t stream) {
  const float* x    = (const float*)d_in[0];
  const float* Wgx  = (const float*)d_in[1];
  const float* bgx  = (const float*)d_in[2];
  const float* Wgh  = (const float*)d_in[3];
  const float* Wix  = (const float*)d_in[4];
  const float* bix  = (const float*)d_in[5];
  const float* Wih  = (const float*)d_in[6];
  const float* Wfx  = (const float*)d_in[7];
  const float* bfx  = (const float*)d_in[8];
  const float* Wfh  = (const float*)d_in[9];
  const float* Wox  = (const float*)d_in[10];
  const float* box_ = (const float*)d_in[11];
  const float* Woh  = (const float*)d_in[12];
  const float* Wout = (const float*)d_in[13];
  const float* bout = (const float*)d_in[14];

  char* ws = (char*)d_ws;
  unsigned short* wpack    = (unsigned short*)(ws + WS_WPACK);
  unsigned short* woutpack = (unsigned short*)(ws + WS_WOUTPACK);
  unsigned short* xpack    = (unsigned short*)(ws + WS_XPACK);
  unsigned short* hpack    = (unsigned short*)(ws + WS_HPACK);
  unsigned*       cnt      = (unsigned*)(ws + WS_CNT);

  init_ws<<<257, 256, 0, stream>>>((unsigned*)hpack, cnt);
  pack_gate_w<<<(4 * 64 * 40 * 512) / 256, 256, 0, stream>>>(
      Wgx, Wgh, Wix, Wih, Wfx, Wfh, Wox, Woh, wpack);
  pack_wout<<<(16 * 32 * 512) / 256, 256, 0, stream>>>(Wout, woutpack);
  pack_x<<<(512 * 4 * 8 * 512) / 256, 256, 0, stream>>>(x, xpack);
  lstm_persistent<<<NWG, 256, 0, stream>>>(wpack, xpack, hpack, bgx, bix, bfx, box_, cnt);
  lstm_out<<<64, 32, 0, stream>>>(hpack, woutpack, bout, (float*)d_out);
}
